// DensityDecoder_43439299232000
// MI455X (gfx1250) — matrix operand order repaired and checked by reading, not compiled, not run on hardware
//
#include <hip/hip_runtime.h>

typedef __attribute__((ext_vector_type(16))) _Float16 v16h;
typedef __attribute__((ext_vector_type(8)))  _Float16 h8;
typedef __attribute__((ext_vector_type(4)))  _Float16 h4;
typedef __attribute__((ext_vector_type(8)))  float    v8f;

union VecH { v16h h; uint4 u[2]; };
union PackH { h8 v; uint4 u; };

#define DIM 256
#define TKV 32
#define NH  8

#define WMMA_F16(A, B, C) \
    __builtin_amdgcn_wmma_f32_16x16x32_f16(false, (A), false, (B), (short)0, (C), false, false)

__device__ __forceinline__ float silu_f(float x) { return x / (1.f + __expf(-x)); }

// Transposed-orientation GEMM:  Y^T = W^T @ X^T  over two 16-row tiles.
//   A operand  = weight fragment (global WT, rows = output features, K contiguous)
//   B operand  = activation fragment (LDS row-major, lane = pair row, K contiguous)
//   D result   = lane holds 8 CONTIGUOUS output features of one pair row
//                -> single packed ds_store_b128 per accumulator.
// B-fragments (2 tiles x KT) preloaded once, reused across all MT output tiles.
template<int KT, int MT, bool SILU>
__device__ __forceinline__ void gemmT(const _Float16* actIn,
                                      const _Float16* __restrict__ WT,
                                      const float* __restrict__ bias,
                                      _Float16* actOut, int lane)
{
    const int n16 = lane & 15;       // pair row within tile (B/D), weight row (A)
    const int hl  = lane >> 4;
    constexpr int K = KT * 32;
    v16h B0[KT], B1[KT];
#pragma unroll
    for (int kt = 0; kt < KT; ++kt) {
        VecH b;
        b.u[0] = *(const uint4*)(actIn + n16 * DIM + kt * 32 + hl * 16);
        b.u[1] = *(const uint4*)(actIn + n16 * DIM + kt * 32 + hl * 16 + 8);
        B0[kt] = b.h;
        b.u[0] = *(const uint4*)(actIn + (16 + n16) * DIM + kt * 32 + hl * 16);
        b.u[1] = *(const uint4*)(actIn + (16 + n16) * DIM + kt * 32 + hl * 16 + 8);
        B1[kt] = b.h;
    }
#pragma unroll
    for (int mt = 0; mt < MT; ++mt) {
        v8f acc0 = {}, acc1 = {};
        const _Float16* wrow = WT + (size_t)(mt * 16 + n16) * K;
#pragma unroll
        for (int kt = 0; kt < KT; ++kt) {
            VecH a;
            a.u[0] = *(const uint4*)(wrow + kt * 32 + hl * 16);
            a.u[1] = *(const uint4*)(wrow + kt * 32 + hl * 16 + 8);
            acc0 = WMMA_F16(a.h, B0[kt], acc0);
            acc1 = WMMA_F16(a.h, B1[kt], acc1);
        }
        const float4* b4 = (const float4*)(bias + mt * 16 + hl * 8);
        float4 bl = b4[0], bh = b4[1];
        float bb[8] = {bl.x, bl.y, bl.z, bl.w, bh.x, bh.y, bh.z, bh.w};
        PackH o0, o1;
#pragma unroll
        for (int r = 0; r < 8; ++r) {
            float x0 = acc0[r] + bb[r];
            float x1 = acc1[r] + bb[r];
            if (SILU) { x0 = silu_f(x0); x1 = silu_f(x1); }
            o0.v[r] = (_Float16)x0;
            o1.v[r] = (_Float16)x1;
        }
        *(uint4*)(actOut + n16 * DIM + mt * 16 + hl * 8)        = o0.u;
        *(uint4*)(actOut + (16 + n16) * DIM + mt * 16 + hl * 8) = o1.u;
    }
}

// ---- prep kernel 1: fp32 (K x N) weight -> f16 transposed (N x K) ----
__global__ void conv_transpose_f16(const float* __restrict__ src, _Float16* __restrict__ dst,
                                   int K, int N) {
    int idx = blockIdx.x * 256 + threadIdx.x;
    if (idx < K * N) {
        int k = idx / N, n = idx - k * N;
        dst[n * K + k] = (_Float16)src[idx];
    }
}

// ---- prep kernel 1b: Wd2 (256 x 2) -> zero-padded transposed f16 (16 x 256) ----
__global__ void wd2_pad_kernel(const float* __restrict__ Wd2, _Float16* __restrict__ dst) {
    int idx = blockIdx.x * 256 + threadIdx.x;
    if (idx < 16 * DIM) {
        int n = idx >> 8, k = idx & 255;
        dst[idx] = (n < 2) ? (_Float16)Wd2[k * 2 + n] : (_Float16)0.f;
    }
}

// ---- prep kernel 2: orbital embedding + 2-layer MLP -> orb (nb x 256 fp32) ----
__global__ void orb_kernel(const int* __restrict__ Z, const int* __restrict__ l,
                           const int* __restrict__ m,
                           const float* __restrict__ elem_tab, const float* __restrict__ l_tab,
                           const float* __restrict__ m_tab,
                           const float* __restrict__ Wp0, const float* __restrict__ bp0,
                           const float* __restrict__ Wp1, const float* __restrict__ bp1,
                           float* __restrict__ orb) {
    __shared__ float emb[64];
    __shared__ float t0[DIM];
    int row = blockIdx.x, t = threadIdx.x;
    if (t < 32)       emb[t] = elem_tab[Z[row] * 32 + t];
    else if (t < 48)  emb[t] = l_tab[l[row] * 16 + (t - 32)];
    else {
        int mi = m[row] + 3; mi = min(max(mi, 0), 4);
        emb[t] = m_tab[mi * 16 + (t - 48)];
    }
    __syncthreads();
    for (int c = t; c < DIM; c += 64) {
        float a = bp0[c];
        for (int k = 0; k < 64; ++k) a += emb[k] * Wp0[k * DIM + c];
        t0[c] = silu_f(a);
    }
    __syncthreads();
    for (int c = t; c < DIM; c += 64) {
        float a = bp1[c];
        for (int k = 0; k < DIM; ++k) a += t0[k] * Wp1[k * DIM + c];
        orb[(size_t)row * DIM + c] = a;
    }
}

// ---- prep kernel 3: LN(latent) -> k (T x D f16) and v^T (D x T f16) ----
__global__ void kv_kernel(const float* __restrict__ latent,
                          const float* __restrict__ gkv, const float* __restrict__ bkv,
                          const float* __restrict__ Wk, const float* __restrict__ bk,
                          const float* __restrict__ Wv, const float* __restrict__ bv,
                          _Float16* __restrict__ k16, _Float16* __restrict__ vT16) {
    __shared__ float kvn[TKV * DIM];
    __shared__ float mu[TKV], rs[TKV];
    int t = threadIdx.x;
    if (t < TKV) {
        float s = 0.f, sq = 0.f;
        for (int c = 0; c < DIM; ++c) { float x = latent[t * DIM + c]; s += x; sq += x * x; }
        float m = s * (1.f / DIM);
        mu[t] = m;
        rs[t] = rsqrtf(sq * (1.f / DIM) - m * m + 1e-5f);
    }
    __syncthreads();
    for (int idx = t; idx < TKV * DIM; idx += 256) {
        int r = idx >> 8, c = idx & 255;
        kvn[idx] = (latent[idx] - mu[r]) * rs[r] * gkv[c] + bkv[c];
    }
    __syncthreads();
    for (int idx = t; idx < TKV * DIM; idx += 256) {
        int r = idx >> 8, c = idx & 255;
        float ak = bk[c], av = bv[c];
        for (int kk = 0; kk < DIM; ++kk) {
            float x = kvn[r * DIM + kk];
            ak += x * Wk[kk * DIM + c];
            av += x * Wv[kk * DIM + c];
        }
        k16[r * DIM + c]  = (_Float16)ak;   // [t][d]: A operand rows for scores (K = d)
        vT16[c * TKV + r] = (_Float16)av;   // [d][t]: A operand rows for attn@v (K = t)
    }
}

// ---- main kernel: 32 pair rows per 32-thread (single-wave) block, 32KB LDS ----
__global__ __launch_bounds__(32) void pair_main(
    const float* __restrict__ orb,
    const _Float16* __restrict__ k16, const _Float16* __restrict__ vT16,
    const _Float16* __restrict__ WqT, const _Float16* __restrict__ WoT,
    const _Float16* __restrict__ Wt0T, const _Float16* __restrict__ Wt1T,
    const _Float16* __restrict__ Wd0T, const _Float16* __restrict__ Wd1T,
    const _Float16* __restrict__ Wd2T,
    const float* __restrict__ ln_gq, const float* __restrict__ ln_bq,
    const float* __restrict__ bqa, const float* __restrict__ bo,
    const float* __restrict__ bt0, const float* __restrict__ bt1,
    const float* __restrict__ bd0, const float* __restrict__ bd1,
    const float* __restrict__ bd2,
    float* __restrict__ outp, int nb, long long P, int spins)
{
    __shared__ __align__(16) _Float16 actA[32 * DIM];
    __shared__ __align__(16) _Float16 actB[32 * DIM];

    const int lane = threadIdx.x;
    const long long tileBase = (long long)blockIdx.x * 32;
    const int n16 = lane & 15;
    const int hl  = lane >> 4;

    // ---- pair rows (orb_i + orb_j) + layernorm; f16 x-cache in-place in actA ----
#pragma unroll
    for (int mb = 0; mb < 2; ++mb) {
        int row = lane >> 1, seg = lane & 1;
        long long pl = tileBase + mb * 16 + row; if (pl >= P) pl = P - 1;
        unsigned p = (unsigned)pl;
        unsigned i = p / (unsigned)nb;
        unsigned j = p - i * (unsigned)nb;
        const float4* oi = (const float4*)(orb + (size_t)i * DIM + seg * 128);
        const float4* oj = (const float4*)(orb + (size_t)j * DIM + seg * 128);
        h4* xc = (h4*)(actA + (mb * 16 + row) * DIM + seg * 128);
        float s = 0.f, sq = 0.f;
#pragma unroll
        for (int c = 0; c < 32; ++c) {
            float4 a = oi[c], b = oj[c];
            float4 x = make_float4(a.x + b.x, a.y + b.y, a.z + b.z, a.w + b.w);
            s  += x.x + x.y + x.z + x.w;
            sq += x.x * x.x + x.y * x.y + x.z * x.z + x.w * x.w;
            h4 xh;
            xh.x = (_Float16)x.x; xh.y = (_Float16)x.y;
            xh.z = (_Float16)x.z; xh.w = (_Float16)x.w;
            xc[c] = xh;
        }
        s  += __shfl_xor(s, 1);
        sq += __shfl_xor(sq, 1);
        float mu   = s * (1.f / DIM);
        float rstd = rsqrtf(sq * (1.f / DIM) - mu * mu + 1e-5f);
        const float4* g4 = (const float4*)(ln_gq + seg * 128);
        const float4* b4 = (const float4*)(ln_bq + seg * 128);
#pragma unroll
        for (int c = 0; c < 32; ++c) {
            h4 xh = xc[c];
            float4 g = g4[c], bb = b4[c];
            h4 o;
            o.x = (_Float16)(((float)xh.x - mu) * rstd * g.x + bb.x);
            o.y = (_Float16)(((float)xh.y - mu) * rstd * g.y + bb.y);
            o.z = (_Float16)(((float)xh.z - mu) * rstd * g.z + bb.z);
            o.w = (_Float16)(((float)xh.w - mu) * rstd * g.w + bb.w);
            xc[c] = o;
        }
    }

    // ---- q = qn @ Wq + bqa -> actB ----
    gemmT<8, 16, false>(actA, WqT, bqa, actB, lane);

    // ---- attention: fully register-resident softmax, per (half, head) ----
    // Score accumulators per lane (hl, n16): s0 -> t = hl*8 + r, s1 -> t = 16 + hl*8 + r.
    // Row n16's 32 scores live across the lane pair (hl=0, hl=1): one shfl_xor(16)
    // for max and sum; normalized probs packed to f16 and exchanged (4 dword shuffles)
    // to assemble the exact B-operand fragment of attn^T for the o-WMMA. No LDS use.
    const float scale = 0.17677669529663687f;  // 1/sqrt(32)
#pragma unroll
    for (int mb = 0; mb < 2; ++mb) {
#pragma unroll
        for (int h = 0; h < NH; ++h) {
            // q B-fragment (K = d)
            VecH bq;
            const _Float16* qrow = actB + (mb * 16 + n16) * DIM + h * 32;
            bq.u[0] = *(const uint4*)(qrow + hl * 16);
            bq.u[1] = *(const uint4*)(qrow + hl * 16 + 8);
            // S^T = K @ Q^T : two t-halves
            v8f s0 = {}, s1 = {};
            {
                VecH a;
                const _Float16* kp = k16 + n16 * DIM + h * 32;
                a.u[0] = *(const uint4*)(kp + hl * 16);
                a.u[1] = *(const uint4*)(kp + hl * 16 + 8);
                s0 = WMMA_F16(a.h, bq.h, s0);
                kp += 16 * DIM;
                a.u[0] = *(const uint4*)(kp + hl * 16);
                a.u[1] = *(const uint4*)(kp + hl * 16 + 8);
                s1 = WMMA_F16(a.h, bq.h, s1);
            }
            // softmax over t (32) for row n16, spread across lane pair (n16, n16+16)
            float lmx = -3.4e38f;
#pragma unroll
            for (int r = 0; r < 8; ++r) lmx = fmaxf(lmx, fmaxf(s0[r], s1[r]));
            float mx = fmaxf(lmx, __shfl_xor(lmx, 16));
            float e0[8], e1[8];
            float lsum = 0.f;
#pragma unroll
            for (int r = 0; r < 8; ++r) {
                e0[r] = __expf((s0[r] - mx) * scale);
                e1[r] = __expf((s1[r] - mx) * scale);
                lsum += e0[r] + e1[r];
            }
            float inv = 1.f / (lsum + __shfl_xor(lsum, 16));
            PackH pk0, pk1;
#pragma unroll
            for (int r = 0; r < 8; ++r) {
                pk0.v[r] = (_Float16)(e0[r] * inv);
                pk1.v[r] = (_Float16)(e1[r] * inv);
            }
            // exchange: hl=0 sends t16-23 (pk1), keeps t0-7 (pk0);
            //           hl=1 sends t8-15  (pk0), keeps t24-31 (pk1)
            uint4 send = hl ? pk0.u : pk1.u;
            uint4 keep = hl ? pk1.u : pk0.u;
            uint4 recv;
            recv.x = (unsigned)__shfl_xor((int)send.x, 16);
            recv.y = (unsigned)__shfl_xor((int)send.y, 16);
            recv.z = (unsigned)__shfl_xor((int)send.z, 16);
            recv.w = (unsigned)__shfl_xor((int)send.w, 16);
            VecH battn;
            battn.u[0] = hl ? recv : keep;   // K elements t = hl*16 + 0..7
            battn.u[1] = hl ? keep : recv;   // K elements t = hl*16 + 8..15
            // O^T = V^T @ Attn^T ; writes replace the just-consumed q columns in-place
#pragma unroll
            for (int mt = 0; mt < 2; ++mt) {
                VecH a;
                const _Float16* vp = vT16 + (h * 32 + mt * 16 + n16) * TKV;
                a.u[0] = *(const uint4*)(vp + hl * 16);
                a.u[1] = *(const uint4*)(vp + hl * 16 + 8);
                v8f acc = {};
                acc = WMMA_F16(a.h, battn.h, acc);
                PackH o;
#pragma unroll
                for (int r = 0; r < 8; ++r) o.v[r] = (_Float16)acc[r];
                *(uint4*)(actB + (mb * 16 + n16) * DIM + h * 32 + mt * 16 + hl * 8) = o.u;
            }
        }
    }

    // ---- remaining projection chain, all WMMA ----
    gemmT<8, 16, false>(actB, WoT,  bo,  actA, lane);   // o @ Wo + bo
    gemmT<8, 16, true >(actA, Wt0T, bt0, actB, lane);   // silu(. @ Wt0 + bt0)
    gemmT<8, 8,  false>(actB, Wt1T, bt1, actA, lane);   // feat = . @ Wt1 + bt1 (32x128)
    gemmT<4, 16, true >(actA, Wd0T, bd0, actB, lane);   // silu(feat @ Wd0 + bd0)
    gemmT<8, 16, true >(actB, Wd1T, bd1, actA, lane);   // silu(. @ Wd1 + bd1)

    // ---- final 256 -> 2 via WMMA (A = zero-padded Wd2T); (re,im) adjacent in acc ----
#pragma unroll
    for (int mb = 0; mb < 2; ++mb) {
        v8f acc = {};
        const _Float16* wrow = Wd2T + (size_t)n16 * DIM;
#pragma unroll
        for (int kt = 0; kt < 8; ++kt) {
            VecH a, b;
            a.u[0] = *(const uint4*)(wrow + kt * 32 + hl * 16);
            a.u[1] = *(const uint4*)(wrow + kt * 32 + hl * 16 + 8);
            const _Float16* arow = actA + (mb * 16 + n16) * DIM + kt * 32;
            b.u[0] = *(const uint4*)(arow + hl * 16);
            b.u[1] = *(const uint4*)(arow + hl * 16 + 8);
            acc = WMMA_F16(a.h, b.h, acc);
        }
        // lanes 0-15 hold out[row n16][m] in acc[m]; m=0 real, m=1 imag
        if (hl == 0) {
            long long p = tileBase + mb * 16 + n16;
            if (p < P) {
                float2 v = make_float2(acc[0] + bd2[0], acc[1] + bd2[1]);
                for (int s = 0; s < spins; ++s)
                    *(float2*)(outp + (size_t)s * (size_t)P * 2 + (size_t)p * 2) = v;
            }
        }
    }
}

extern "C" void kernel_launch(void* const* d_in, const int* in_sizes, int n_in,
                              void* d_out, int out_size, void* d_ws, size_t ws_size,
                              hipStream_t stream) {
    const float* latent   = (const float*)d_in[0];
    const int*   Z        = (const int*)d_in[1];
    const int*   l        = (const int*)d_in[2];
    const int*   m        = (const int*)d_in[3];
    const float* elem_tab = (const float*)d_in[4];
    const float* l_tab    = (const float*)d_in[5];
    const float* m_tab    = (const float*)d_in[6];
    const float* Wp0 = (const float*)d_in[7],  *bp0 = (const float*)d_in[8];
    const float* Wp1 = (const float*)d_in[9],  *bp1 = (const float*)d_in[10];
    const float* ln_gq = (const float*)d_in[11], *ln_bq = (const float*)d_in[12];
    const float* ln_gkv = (const float*)d_in[13], *ln_bkv = (const float*)d_in[14];
    const float* Wq = (const float*)d_in[15], *bqa = (const float*)d_in[16];
    const float* Wk = (const float*)d_in[17], *bk  = (const float*)d_in[18];
    const float* Wv = (const float*)d_in[19], *bv  = (const float*)d_in[20];
    const float* Wo = (const float*)d_in[21], *bo  = (const float*)d_in[22];
    const float* Wt0 = (const float*)d_in[23], *bt0 = (const float*)d_in[24];
    const float* Wt1 = (const float*)d_in[25], *bt1 = (const float*)d_in[26];
    const float* Wd0 = (const float*)d_in[27], *bd0 = (const float*)d_in[28];
    const float* Wd1 = (const float*)d_in[29], *bd1 = (const float*)d_in[30];
    const float* Wd2 = (const float*)d_in[31], *bd2 = (const float*)d_in[32];

    const int nb = in_sizes[1];
    const long long P = (long long)nb * nb;
    int r = (int)((long long)out_size / P);
    int spins = (r >= 2) ? (r / 2) : 1;   // out_size counts floats (complex64 -> 2 floats)

    // workspace layout
    char* ws = (char*)d_ws;
    size_t off = 0;
    float*    orb  = (float*)(ws + off);    off += (size_t)nb * DIM * sizeof(float);
    _Float16* k16  = (_Float16*)(ws + off); off += (size_t)TKV * DIM * 2;
    _Float16* vT16 = (_Float16*)(ws + off); off += (size_t)DIM * TKV * 2;
    _Float16* WqT  = (_Float16*)(ws + off); off += (size_t)DIM * DIM * 2;
    _Float16* WoT  = (_Float16*)(ws + off); off += (size_t)DIM * DIM * 2;
    _Float16* Wt0T = (_Float16*)(ws + off); off += (size_t)DIM * DIM * 2;
    _Float16* Wt1T = (_Float16*)(ws + off); off += (size_t)DIM * (DIM / 2) * 2;
    _Float16* Wd0T = (_Float16*)(ws + off); off += (size_t)(DIM / 2) * DIM * 2;
    _Float16* Wd1T = (_Float16*)(ws + off); off += (size_t)DIM * DIM * 2;
    _Float16* Wd2T = (_Float16*)(ws + off); off += (size_t)16 * DIM * 2;
    (void)off; (void)ws_size; (void)n_in;

    auto gcv = [](int K, int N) { return dim3((unsigned)((K * N + 255) / 256)); };
    conv_transpose_f16<<<gcv(DIM, DIM), 256, 0, stream>>>(Wq,  WqT,  DIM, DIM);
    conv_transpose_f16<<<gcv(DIM, DIM), 256, 0, stream>>>(Wo,  WoT,  DIM, DIM);
    conv_transpose_f16<<<gcv(DIM, DIM), 256, 0, stream>>>(Wt0, Wt0T, DIM, DIM);
    conv_transpose_f16<<<gcv(DIM, DIM / 2), 256, 0, stream>>>(Wt1, Wt1T, DIM, DIM / 2);
    conv_transpose_f16<<<gcv(DIM / 2, DIM), 256, 0, stream>>>(Wd0, Wd0T, DIM / 2, DIM);
    conv_transpose_f16<<<gcv(DIM, DIM), 256, 0, stream>>>(Wd1, Wd1T, DIM, DIM);
    wd2_pad_kernel<<<dim3((16 * DIM + 255) / 256), 256, 0, stream>>>(Wd2, Wd2T);

    orb_kernel<<<dim3((unsigned)nb), 64, 0, stream>>>(Z, l, m, elem_tab, l_tab, m_tab,
                                                      Wp0, bp0, Wp1, bp1, orb);
    kv_kernel<<<dim3(1), 256, 0, stream>>>(latent, ln_gkv, ln_bkv, Wk, bk, Wv, bv, k16, vT16);

    unsigned nTiles = (unsigned)((P + 31) / 32);
    pair_main<<<dim3(nTiles), 32, 0, stream>>>(
        orb, k16, vT16, WqT, WoT, Wt0T, Wt1T, Wd0T, Wd1T, Wd2T,
        ln_gq, ln_bq, bqa, bo, bt0, bt1, bd0, bd1, bd2,
        (float*)d_out, nb, P, spins);
}
